// SanaMultiscaleLinearAttention_62947040690299
// MI455X (gfx1250) — matrix operand order repaired and checked by reading, not compiled, not run on hardware
//
#include <hip/hip_runtime.h>
#include <hip/hip_bf16.h>

typedef __attribute__((ext_vector_type(16))) __bf16 v16bf;
typedef __attribute__((ext_vector_type(8)))  __bf16 v8bf;
typedef __attribute__((ext_vector_type(8)))  float  v8f;

#define GN 16384            // H*W per image
#define NB 2                // batch
#define QKV_CH 1536
#define ATT_CH 1024
#define OUTC 512
#define NGRP 128            // head-groups per batch
#define NCHUNK 8            // N-splits for attention kernels

union FragU  { v16bf v; v8bf h[2]; };
union Pack8  { __bf16 b[8]; uint4 u; };   // 8 bf16 -> one ds_store_b128
union Pack4  { __bf16 b[4]; uint2 u; };   // 4 bf16 -> one ds_store_b64

// ---------------------------------------------------------------------------
// Tiled bf16 WMMA GEMM:  Y[b][m][n] = sum_k A[m][k] * X[b][k][n]  (+ fused BN)
// Block tile 128(M) x 128(N), K-step 32. 8 waves: 2(M)x4(N), each 64x32.
// ---------------------------------------------------------------------------
template<int K, bool SPLIT3, bool FUSE_BN>
__global__ __launch_bounds__(256) void gemm_bf16_k(
    const float* __restrict__ A0, const float* __restrict__ A1,
    const float* __restrict__ A2,
    const float* __restrict__ X, float* __restrict__ Y,
    const float* __restrict__ bng, const float* __restrict__ bnb,
    const float* __restrict__ bnm, const float* __restrict__ bnv)
{
  const int tid    = threadIdx.x;
  const int lane   = tid & 31;
  const int wave   = tid >> 5;
  const int laneHi = lane >> 4;       // 0/1
  const int laneLo = lane & 15;
  const int nBase  = blockIdx.x * 128;
  const int mBase  = blockIdx.y * 128;
  const int b      = blockIdx.z;
  const int M      = gridDim.y * 128;

  const float* Xb = X + (size_t)b * K * GN;
  float*       Yb = Y + (size_t)b * M * GN;

  __shared__ __bf16 lsA[128 * 40];    // [m][k], padded stride 40
  __shared__ __bf16 lsB[128 * 40];    // [n][k] (transposed), padded stride 40

  // A staging: 2 threads per row, each converts 16 consecutive k floats
  const int a_row = tid >> 1;
  const int a_kh  = (tid & 1) * 16;
  const int a_gr  = mBase + a_row;
  const float* arow;
  if (SPLIT3) {
    arow = (a_gr < 512)  ? A0 + (size_t)a_gr * K
         : (a_gr < 1024) ? A1 + (size_t)(a_gr - 512) * K
                         : A2 + (size_t)(a_gr - 1024) * K;
  } else {
    arow = A0 + (size_t)a_gr * K;
  }
  // B staging: thread owns a 4(k) x 4(n) micro-tile
  const int b_n0 = (tid & 31) * 4;    // 0..124
  const int b_k0 = (tid >> 5) * 4;    // 0..28

  const int waveM = wave >> 2;        // 0..1 -> 64 rows each
  const int waveN = wave & 3;         // 0..3 -> 32 cols each

  v8f acc[4][2];
  const v8f vzero = {0.f,0.f,0.f,0.f,0.f,0.f,0.f,0.f};
#pragma unroll
  for (int im = 0; im < 4; ++im)
#pragma unroll
    for (int in = 0; in < 2; ++in) acc[im][in] = vzero;

  for (int k0 = 0; k0 < K; k0 += 32) {
    __syncthreads();
    // prefetch next K slice (global_prefetch_b8)
    if (k0 + 32 < K) {
      __builtin_prefetch(arow + (k0 + 32) + a_kh, 0, 1);
      __builtin_prefetch(Xb + (size_t)(k0 + 32 + b_k0) * GN + nBase + b_n0, 0, 1);
    }
    // ---- stage A (row-major, fp32 -> bf16, packed b128 stores) ----
    {
      const float4* s = (const float4*)(arow + k0 + a_kh);
      float4 f0 = s[0], f1 = s[1], f2 = s[2], f3 = s[3];
      Pack8 p0, p1;
      p0.b[0]=(__bf16)f0.x; p0.b[1]=(__bf16)f0.y; p0.b[2]=(__bf16)f0.z; p0.b[3]=(__bf16)f0.w;
      p0.b[4]=(__bf16)f1.x; p0.b[5]=(__bf16)f1.y; p0.b[6]=(__bf16)f1.z; p0.b[7]=(__bf16)f1.w;
      p1.b[0]=(__bf16)f2.x; p1.b[1]=(__bf16)f2.y; p1.b[2]=(__bf16)f2.z; p1.b[3]=(__bf16)f2.w;
      p1.b[4]=(__bf16)f3.x; p1.b[5]=(__bf16)f3.y; p1.b[6]=(__bf16)f3.z; p1.b[7]=(__bf16)f3.w;
      *(uint4*)&lsA[a_row * 40 + a_kh]     = p0.u;
      *(uint4*)&lsA[a_row * 40 + a_kh + 8] = p1.u;
    }
    // ---- stage B transposed: lsB[n][k], packed b64 stores ----
    {
      const float* src = Xb + (size_t)(k0 + b_k0) * GN + nBase + b_n0;
      float4 r0 = *(const float4*)(src);
      float4 r1 = *(const float4*)(src + GN);
      float4 r2 = *(const float4*)(src + 2 * GN);
      float4 r3 = *(const float4*)(src + 3 * GN);
      Pack4 q0, q1, q2, q3;
      q0.b[0]=(__bf16)r0.x; q0.b[1]=(__bf16)r1.x; q0.b[2]=(__bf16)r2.x; q0.b[3]=(__bf16)r3.x;
      q1.b[0]=(__bf16)r0.y; q1.b[1]=(__bf16)r1.y; q1.b[2]=(__bf16)r2.y; q1.b[3]=(__bf16)r3.y;
      q2.b[0]=(__bf16)r0.z; q2.b[1]=(__bf16)r1.z; q2.b[2]=(__bf16)r2.z; q2.b[3]=(__bf16)r3.z;
      q3.b[0]=(__bf16)r0.w; q3.b[1]=(__bf16)r1.w; q3.b[2]=(__bf16)r2.w; q3.b[3]=(__bf16)r3.w;
      *(uint2*)&lsB[(b_n0 + 0) * 40 + b_k0] = q0.u;
      *(uint2*)&lsB[(b_n0 + 1) * 40 + b_k0] = q1.u;
      *(uint2*)&lsB[(b_n0 + 2) * 40 + b_k0] = q2.u;
      *(uint2*)&lsB[(b_n0 + 3) * 40 + b_k0] = q3.u;
    }
    __syncthreads();

    // ---- fragments ----
    v16bf af[4], bf_[2];
#pragma unroll
    for (int im = 0; im < 4; ++im) {
      // A layout (16-bit, 16x32): elems 0..7 -> K = laneHi*8 + i,
      //                           elems 8..15 -> K = 16 + laneHi*8 + (i-8)
      const __bf16* p = &lsA[(waveM * 64 + im * 16 + laneLo) * 40];
      FragU t;
      t.h[0] = *(const v8bf*)(p + laneHi * 8);
      t.h[1] = *(const v8bf*)(p + 16 + laneHi * 8);
      af[im] = t.v;
    }
#pragma unroll
    for (int in = 0; in < 2; ++in) {
      // B layout (16-bit, 32x16): col = laneLo, elems 0..7 -> K = laneHi*8 + i,
      //                           elems 8..15 -> K = 16 + laneHi*8 + (i-8)
      const __bf16* p = &lsB[(waveN * 32 + in * 16 + laneLo) * 40 + laneHi * 8];
      FragU t;
      t.h[0] = *(const v8bf*)(p);
      t.h[1] = *(const v8bf*)(p + 16);
      bf_[in] = t.v;
    }
#pragma unroll
    for (int im = 0; im < 4; ++im)
#pragma unroll
      for (int in = 0; in < 2; ++in)
        acc[im][in] = __builtin_amdgcn_wmma_f32_16x16x32_bf16(
            false, af[im], false, bf_[in], (short)0, acc[im][in], false, false);
  }

  // ---- store (C/D layout: vgpr r -> M=r(+8 for lanes>=16), lane -> N) ----
  const int addM = laneHi * 8;
#pragma unroll
  for (int im = 0; im < 4; ++im) {
#pragma unroll
    for (int in = 0; in < 2; ++in) {
      const int row0 = mBase + waveM * 64 + im * 16 + addM;
      const int col  = nBase + waveN * 32 + in * 16 + laneLo;
#pragma unroll
      for (int r = 0; r < 8; ++r) {
        const int row = row0 + r;
        float v = acc[im][in][r];
        if (FUSE_BN) {
          const float inv = bng[row] * rsqrtf(bnv[row] + 1e-5f);
          v = v * inv + (bnb[row] - bnm[row] * inv);
        }
        Yb[(size_t)row * GN + col] = v;
      }
    }
  }
}

// ---------------------------------------------------------------------------
// Fused multiscale branch: depthwise 5x5 (SAME) + grouped 1x1 (8 in -> 8 out)
// ---------------------------------------------------------------------------
__global__ __launch_bounds__(256) void msconv_kernel(
    const float* __restrict__ qkv, const float* __restrict__ dw,
    const float* __restrict__ pw, float* __restrict__ ms)
{
  const size_t idx = (size_t)blockIdx.x * 256 + threadIdx.x;
  const size_t total = (size_t)NB * 192 * GN;
  if (idx >= total) return;
  const int n = (int)(idx % GN);
  const int g = (int)((idx / GN) % 192);
  const int b = (int)(idx / ((size_t)GN * 192));
  const int y = n >> 7, x = n & 127;

  const float* src = qkv + ((size_t)b * QKV_CH + g * 8) * GN;
  float acc[8] = {0.f,0.f,0.f,0.f,0.f,0.f,0.f,0.f};

#pragma unroll
  for (int dy = -2; dy <= 2; ++dy) {
    const int yy = y + dy;
    if (yy < 0 || yy > 127) continue;
#pragma unroll
    for (int dx = -2; dx <= 2; ++dx) {
      const int xx = x + dx;
      if (xx < 0 || xx > 127) continue;
      const int off = (yy << 7) + xx;
      const int w   = (dy + 2) * 5 + (dx + 2);
#pragma unroll
      for (int i = 0; i < 8; ++i)
        acc[i] += src[(size_t)i * GN + off] * dw[(g * 8 + i) * 25 + w];
    }
  }
#pragma unroll
  for (int j = 0; j < 8; ++j) {
    float o = 0.f;
#pragma unroll
    for (int i = 0; i < 8; ++i) o += pw[(g * 8 + j) * 8 + i] * acc[i];
    ms[((size_t)b * QKV_CH + g * 8 + j) * GN + n] = o;
  }
}

// ---------------------------------------------------------------------------
// Linear attention, pass 1: scores[bg][d*8+e] += sum_n v_pad[d,n]*relu(k[e,n])
// grid = NB*NGRP*NCHUNK blocks; wave32 shfl reduce -> LDS -> global atomics
// ---------------------------------------------------------------------------
__device__ __forceinline__ const float* grp_base(
    const float* qkv, const float* ms, int b, int g)
{
  return (g < 64) ? (qkv + ((size_t)b * QKV_CH + 24 * g) * GN)
                  : (ms  + ((size_t)b * QKV_CH + 24 * (g - 64)) * GN);
}

__global__ __launch_bounds__(256) void attn_scores_kernel(
    const float* __restrict__ qkv, const float* __restrict__ ms,
    float* __restrict__ scores)
{
  const int blk   = blockIdx.x;
  const int chunk = blk & (NCHUNK - 1);
  const int bg    = blk >> 3;
  const int b     = bg >> 7;
  const int g     = bg & (NGRP - 1);
  const int tid   = threadIdx.x;
  const int lane  = tid & 31;
  const int span  = GN / NCHUNK;      // 2048
  const int n0    = chunk * span;

  const float* base = grp_base(qkv, ms, b, g);

  float s72[72];
#pragma unroll
  for (int i = 0; i < 72; ++i) s72[i] = 0.f;

  for (int n = n0 + tid; n < n0 + span; n += 256) {
    float kv[8], vv[8];
#pragma unroll
    for (int e = 0; e < 8; ++e) kv[e] = fmaxf(base[(size_t)(8 + e) * GN + n], 0.f);
#pragma unroll
    for (int d = 0; d < 8; ++d) vv[d] = base[(size_t)(16 + d) * GN + n];
#pragma unroll
    for (int d = 0; d < 8; ++d)
#pragma unroll
      for (int e = 0; e < 8; ++e) s72[d * 8 + e] += vv[d] * kv[e];
#pragma unroll
    for (int e = 0; e < 8; ++e) s72[64 + e] += kv[e];
  }

  __shared__ float sc[72];
  for (int i = tid; i < 72; i += 256) sc[i] = 0.f;
  __syncthreads();

#pragma unroll
  for (int i = 0; i < 72; ++i) {
    float v = s72[i];
    v += __shfl_xor(v, 16, 32);
    v += __shfl_xor(v,  8, 32);
    v += __shfl_xor(v,  4, 32);
    v += __shfl_xor(v,  2, 32);
    v += __shfl_xor(v,  1, 32);
    if (lane == 0) atomicAdd(&sc[i], v);
  }
  __syncthreads();

  if (tid < 72) atomicAdd(&scores[(size_t)bg * 72 + tid], sc[tid]);
}

// ---------------------------------------------------------------------------
// Linear attention, pass 2: out[d][n] = sum_e scores[d][e]*relu(q[e,n]),
// final = out[0:8]/(out[8]+eps)
// ---------------------------------------------------------------------------
__global__ __launch_bounds__(256) void attn_apply_kernel(
    const float* __restrict__ qkv, const float* __restrict__ ms,
    const float* __restrict__ scores, float* __restrict__ att)
{
  const int blk   = blockIdx.x;
  const int chunk = blk & (NCHUNK - 1);
  const int bg    = blk >> 3;
  const int b     = bg >> 7;
  const int g     = bg & (NGRP - 1);
  const int tid   = threadIdx.x;
  const int span  = GN / NCHUNK;      // 2048
  const int n0    = chunk * span;

  const float* base = grp_base(qkv, ms, b, g);

  __shared__ float sc[72];
  for (int i = tid; i < 72; i += 256) sc[i] = scores[(size_t)bg * 72 + i];
  __syncthreads();

  float sl[72];
#pragma unroll
  for (int i = 0; i < 72; ++i) sl[i] = sc[i];

  float* dst = att + ((size_t)b * ATT_CH + g * 8) * GN;
  for (int n = n0 + tid; n < n0 + span; n += 256) {
    float qv[8];
#pragma unroll
    for (int e = 0; e < 8; ++e) qv[e] = fmaxf(base[(size_t)e * GN + n], 0.f);
    float den = 1e-15f;
#pragma unroll
    for (int e = 0; e < 8; ++e) den += sl[64 + e] * qv[e];
    const float rden = 1.f / den;
#pragma unroll
    for (int d = 0; d < 8; ++d) {
      float num = 0.f;
#pragma unroll
      for (int e = 0; e < 8; ++e) num += sl[d * 8 + e] * qv[e];
      dst[(size_t)d * GN + n] = num * rden;
    }
  }
}

// ---------------------------------------------------------------------------
extern "C" void kernel_launch(void* const* d_in, const int* in_sizes, int n_in,
                              void* d_out, int out_size, void* d_ws, size_t ws_size,
                              hipStream_t stream) {
  (void)in_sizes; (void)n_in; (void)out_size; (void)ws_size;
  const float* x    = (const float*)d_in[0];
  const float* Wq   = (const float*)d_in[1];
  const float* Wk   = (const float*)d_in[2];
  const float* Wv   = (const float*)d_in[3];
  const float* dw   = (const float*)d_in[4];
  const float* pw   = (const float*)d_in[5];
  const float* Wout = (const float*)d_in[6];
  const float* bng  = (const float*)d_in[7];
  const float* bnb  = (const float*)d_in[8];
  const float* bnm  = (const float*)d_in[9];
  const float* bnv  = (const float*)d_in[10];
  float* out = (float*)d_out;

  float* qkv    = (float*)d_ws;                               // B*1536*GN
  float* msb    = qkv + (size_t)NB * QKV_CH * GN;             // B*1536*GN
  float* att    = msb + (size_t)NB * QKV_CH * GN;             // B*1024*GN
  float* scores = att + (size_t)NB * ATT_CH * GN;             // B*128*72

  dim3 blk(256);

  // 1) fused QKV projection GEMM: (1536x512) x (512x16384) per batch
  gemm_bf16_k<512, true, false><<<dim3(GN / 128, QKV_CH / 128, NB), blk, 0, stream>>>(
      Wq, Wk, Wv, x, qkv, nullptr, nullptr, nullptr, nullptr);

  // 2) fused depthwise 5x5 + grouped 1x1 multiscale branch
  {
    const size_t total = (size_t)NB * 192 * GN;
    msconv_kernel<<<(unsigned)((total + 255) / 256), blk, 0, stream>>>(qkv, dw, pw, msb);
  }

  // 3) linear attention (scores reduce, then apply)
  hipMemsetAsync(scores, 0, (size_t)NB * NGRP * 72 * sizeof(float), stream);
  attn_scores_kernel<<<NB * NGRP * NCHUNK, blk, 0, stream>>>(qkv, msb, scores);
  attn_apply_kernel<<<NB * NGRP * NCHUNK, blk, 0, stream>>>(qkv, msb, scores, att);

  // 4) to_out GEMM (512x1024) x (1024x16384) with fused eval-mode BatchNorm
  gemm_bf16_k<1024, false, true><<<dim3(GN / 128, OUTC / 128, NB), blk, 0, stream>>>(
      Wout, nullptr, nullptr, att, out, bng, bnb, bnm, bnv);
}
